// Olmo3MoeSparseMLP_23141283791732
// MI455X (gfx1250) — matrix operand order repaired and checked by reading, not compiled, not run on hardware
//
#include <hip/hip_runtime.h>
#include <hip/hip_bf16.h>
#include <math.h>

// Problem sizes (fixed by the reference)
#define Bq 4
#define Sq 2048
#define Hh 2048
#define Ee 32
#define Kk 4
#define Ii 1024
#define Nn (Bq * Sq)   // 8192 tokens

typedef __attribute__((ext_vector_type(16))) __bf16 v16bf;
typedef __attribute__((ext_vector_type(8)))  float  v8f;

#define WMMA_BF16(a, b, c) \
    __builtin_amdgcn_wmma_f32_16x16x32_bf16(false, (a), false, (b), (short)0, (c), false, false)

union FragU {
    v16bf v;
    uint4 q[2];
};

__device__ __forceinline__ v16bf pack16(const float* f) {
    v16bf r;
#pragma unroll
    for (int i = 0; i < 16; ++i) r[i] = (__bf16)f[i];
    return r;
}

// ---- A fragment (16x32 bf16) built from a global fp32 row ---------------
// ISA layout: lanes 0-15  -> M=lane,    K = {0..7, 16..23}
//             lanes 16-31 -> M=lane-16, K = {8..15, 24..31}
__device__ __forceinline__ v16bf load_a_row(const float* __restrict__ row, int kb0) {
    float f[16];
    const float4 a0 = *(const float4*)(row + kb0);
    const float4 a1 = *(const float4*)(row + kb0 + 4);
    const float4 a2 = *(const float4*)(row + kb0 + 16);
    const float4 a3 = *(const float4*)(row + kb0 + 20);
    f[0]=a0.x;  f[1]=a0.y;  f[2]=a0.z;  f[3]=a0.w;
    f[4]=a1.x;  f[5]=a1.y;  f[6]=a1.z;  f[7]=a1.w;
    f[8]=a2.x;  f[9]=a2.y;  f[10]=a2.z; f[11]=a2.w;
    f[12]=a3.x; f[13]=a3.y; f[14]=a3.z; f[15]=a3.w;
    return pack16(f);
}

// ---- B fragment (32x16 bf16) from pre-transposed bf16 weights -----------
// Wt row = output column, K contiguous: two 16-byte loads, no conversion.
__device__ __forceinline__ v16bf load_b_t(const __bf16* __restrict__ p) {
    FragU u;
    u.q[0] = *(const uint4*)(p);
    u.q[1] = *(const uint4*)(p + 16);
    return u.v;
}

// ---- A fragment from LDS (bf16 row-major [16][K]) -----------------------
__device__ __forceinline__ v16bf load_a_lds(const __bf16* hrow, int kb0) {
    FragU u;
    u.q[0] = *(const uint4*)(hrow + kb0);
    u.q[1] = *(const uint4*)(hrow + kb0 + 16);
    return u.v;
}

// =========================================================================
// Prep kernel: tiled transpose + fp32 -> bf16 convert.
//   src: [E][R][C] fp32  ->  dst: [E][C][R] bf16       grid (C/32, R/32, E)
// =========================================================================
__global__ __launch_bounds__(256) void transpose_cvt_kernel(
    const float* __restrict__ src, __bf16* __restrict__ dst, int R, int C)
{
    __shared__ __bf16 tile[32][33];
    const int e = blockIdx.z;
    const float* s = src + (size_t)e * R * C;
    __bf16*      d = dst + (size_t)e * R * C;
    const int c0 = blockIdx.x * 32;
    const int r0 = blockIdx.y * 32;
    const int tx = threadIdx.x & 31;
    const int ty = threadIdx.x >> 5;

#pragma unroll
    for (int p = 0; p < 4; ++p) {
        const int r = ty + p * 8;
        tile[r][tx] = (__bf16)s[(size_t)(r0 + r) * C + c0 + tx];
    }
    __syncthreads();
#pragma unroll
    for (int p = 0; p < 4; ++p) {
        const int rr = ty + p * 8;
        d[(size_t)(c0 + rr) * R + r0 + tx] = tile[tx][rr];
    }
}

// =========================================================================
// Kernel 1: router — logits, softmax, top-4, compacted per-expert lists
// =========================================================================
__global__ __launch_bounds__(256) void router_kernel(
    const float* __restrict__ x, const float* __restrict__ Wgate,
    int* __restrict__ counts, int* __restrict__ tok_ids, float* __restrict__ tok_w)
{
    const int lane = threadIdx.x & 31;
    const int wv   = threadIdx.x >> 5;
    const int t    = blockIdx.x * 8 + wv;
    if (t >= Nn) return;

    const float* xr = x + (size_t)t * Hh;
    const float* wr = Wgate + (size_t)lane * Hh;   // lane == expert
    float acc = 0.0f;
    for (int h = 0; h < Hh; h += 4) {
        const float4 xv = *(const float4*)(xr + h);
        const float4 gv = *(const float4*)(wr + h);
        acc += xv.x * gv.x + xv.y * gv.y + xv.z * gv.z + xv.w * gv.w;
    }

    float m = acc;
#pragma unroll
    for (int off = 16; off >= 1; off >>= 1) m = fmaxf(m, __shfl_xor(m, off, 32));
    const float p = __expf(acc - m);
    float s = p;
#pragma unroll
    for (int off = 16; off >= 1; off >>= 1) s += __shfl_xor(s, off, 32);
    const float score = p / s;

    float sv = score;
    for (int it = 0; it < Kk; ++it) {
        float v = sv;
        int   idx = lane;
#pragma unroll
        for (int off = 16; off >= 1; off >>= 1) {
            const float ov = __shfl_xor(v, off, 32);
            const int   oi = __shfl_xor(idx, off, 32);
            if (ov > v || (ov == v && oi < idx)) { v = ov; idx = oi; }
        }
        if (lane == idx) {
            const int slot = atomicAdd(&counts[lane], 1);
            tok_ids[lane * Nn + slot] = t;
            tok_w[lane * Nn + slot]   = score;
            sv = -__builtin_inff();
        }
    }
}

// =========================================================================
// Kernel 2: fused SwiGLU expert MLP, software-pipelined WMMA main loops.
// Block = 256 threads (8 waves), tile = 16 tokens of one expert.
// =========================================================================
__global__ __launch_bounds__(256) void moe_expert_kernel(
    const float* __restrict__ x,
    const __bf16* __restrict__ WgT,   // [E][I][H] bf16
    const __bf16* __restrict__ WuT,   // [E][I][H] bf16
    const __bf16* __restrict__ WdT,   // [E][H][I] bf16
    const int* __restrict__ counts, const int* __restrict__ tok_ids,
    const float* __restrict__ tok_w, float* __restrict__ out)
{
    __shared__ __bf16 Hs[16][Ii];       // 32 KB intermediate activations
    __shared__ int    sm_tok[16];
    __shared__ float  sm_w[16];

    const int e   = blockIdx.y;
    const int cnt = counts[e];
    const int m0  = blockIdx.x * 16;
    if (m0 >= cnt) return;

    const int tid   = threadIdx.x;
    const int lane  = tid & 31;
    const int wv    = tid >> 5;
    const int mlane = lane & 15;
    const int hi    = lane >> 4;

    if (tid < 16) {
        const int idx = m0 + tid;
        if (idx < cnt) { sm_tok[tid] = tok_ids[e * Nn + idx]; sm_w[tid] = tok_w[e * Nn + idx]; }
        else           { sm_tok[tid] = tok_ids[e * Nn + m0];  sm_w[tid] = 0.0f; }
    }
    __syncthreads();

    const float* xrow = x + (size_t)sm_tok[mlane] * Hh;
    const v8f zero = {0.f, 0.f, 0.f, 0.f, 0.f, 0.f, 0.f, 0.f};

    // ================= Phase 1: gate/up projections ======================
    // 2 passes x 4 column tiles per wave; 1-deep software pipeline on B+A.
#pragma unroll 1
    for (int pass = 0; pass < 2; ++pass) {
        const size_t colb = (size_t)((wv * 8 + pass * 4) * 16 + mlane);
        const __bf16* pgb = WgT + ((size_t)e * Ii + colb) * Hh + hi * 8;
        const __bf16* pub = WuT + ((size_t)e * Ii + colb) * Hh + hi * 8;

        v8f ag[4], au[4];
#pragma unroll
        for (int t = 0; t < 4; ++t) { ag[t] = zero; au[t] = zero; }

        v16bf bg0[4], bu0[4], bg1[4], bu1[4];
        v16bf af0, af1;

        // prologue: buffer 0 @ k = 0
#pragma unroll
        for (int t = 0; t < 4; ++t) {
            bg0[t] = load_b_t(pgb + (size_t)t * 16 * Hh);
            bu0[t] = load_b_t(pub + (size_t)t * 16 * Hh);
        }
        af0 = load_a_row(xrow, hi * 8);

#pragma unroll 1
        for (int kc2 = 0; kc2 < Hh / 32; kc2 += 2) {
            // prefetch buffer 1 @ k = kc2+1 (always in range)
            const int k1 = (kc2 + 1) * 32;
#pragma unroll
            for (int t = 0; t < 4; ++t) {
                bg1[t] = load_b_t(pgb + (size_t)t * 16 * Hh + k1);
                bu1[t] = load_b_t(pub + (size_t)t * 16 * Hh + k1);
            }
            af1 = load_a_row(xrow, k1 + hi * 8);

            // compute with buffer 0 (loads issued one iteration ago)
#pragma unroll
            for (int t = 0; t < 4; ++t) {
                ag[t] = WMMA_BF16(af0, bg0[t], ag[t]);
                au[t] = WMMA_BF16(af0, bu0[t], au[t]);
            }

            // prefetch buffer 0 @ k = kc2+2
            if (kc2 + 2 < Hh / 32) {
                const int k2 = (kc2 + 2) * 32;
#pragma unroll
                for (int t = 0; t < 4; ++t) {
                    bg0[t] = load_b_t(pgb + (size_t)t * 16 * Hh + k2);
                    bu0[t] = load_b_t(pub + (size_t)t * 16 * Hh + k2);
                }
                af0 = load_a_row(xrow, k2 + hi * 8);
            }

            // compute with buffer 1
#pragma unroll
            for (int t = 0; t < 4; ++t) {
                ag[t] = WMMA_BF16(af1, bg1[t], ag[t]);
                au[t] = WMMA_BF16(af1, bu1[t], au[t]);
            }
        }

        // silu(gate) * up -> LDS (bf16); C layout: lane<16 M=r, else M=r+8
#pragma unroll
        for (int t = 0; t < 4; ++t) {
            const int jb = (wv * 8 + pass * 4 + t) * 16;
#pragma unroll
            for (int r = 0; r < 8; ++r) {
                const int M = r + hi * 8;
                const float g = ag[t][r];
                const float u = au[t][r];
                const float h = (g / (1.0f + __expf(-g))) * u;
                Hs[M][jb + mlane] = (__bf16)h;
            }
        }
    }
    __syncthreads();

    // ================= Phase 2: down projection + combine ================
    // 2 passes x 8 output tiles per wave; pipelined B (global) + A (LDS).
    const __bf16* hrow = &Hs[mlane][0];
#pragma unroll 1
    for (int pass = 0; pass < 2; ++pass) {
        const int nb0 = wv * 256 + pass * 128;
        const __bf16* pdb = WdT + ((size_t)e * Hh + nb0 + mlane) * Ii + hi * 8;

        v8f acc[8];
#pragma unroll
        for (int t = 0; t < 8; ++t) acc[t] = zero;

        v16bf bd0[8], bd1[8];
        v16bf af0, af1;

#pragma unroll
        for (int t = 0; t < 8; ++t) bd0[t] = load_b_t(pdb + (size_t)t * 16 * Ii);
        af0 = load_a_lds(hrow, hi * 8);

#pragma unroll 1
        for (int kc2 = 0; kc2 < Ii / 32; kc2 += 2) {
            const int k1 = (kc2 + 1) * 32;
#pragma unroll
            for (int t = 0; t < 8; ++t) bd1[t] = load_b_t(pdb + (size_t)t * 16 * Ii + k1);
            af1 = load_a_lds(hrow, k1 + hi * 8);

#pragma unroll
            for (int t = 0; t < 8; ++t) acc[t] = WMMA_BF16(af0, bd0[t], acc[t]);

            if (kc2 + 2 < Ii / 32) {
                const int k2 = (kc2 + 2) * 32;
#pragma unroll
                for (int t = 0; t < 8; ++t) bd0[t] = load_b_t(pdb + (size_t)t * 16 * Ii + k2);
                af0 = load_a_lds(hrow, k2 + hi * 8);
            }

#pragma unroll
            for (int t = 0; t < 8; ++t) acc[t] = WMMA_BF16(af1, bd1[t], acc[t]);
        }

        // scale by routing weight; K-way combine via fp32 atomics
#pragma unroll
        for (int t = 0; t < 8; ++t) {
            const int nb = nb0 + t * 16;
#pragma unroll
            for (int r = 0; r < 8; ++r) {
                const int M = r + hi * 8;
                if (m0 + M < cnt) {
                    const float val = acc[t][r] * sm_w[M];
                    atomicAdd(&out[(size_t)sm_tok[M] * Hh + nb + mlane], val);
                }
            }
        }
    }
}

// =========================================================================
extern "C" void kernel_launch(void* const* d_in, const int* in_sizes, int n_in,
                              void* d_out, int out_size, void* d_ws, size_t ws_size,
                              hipStream_t stream) {
    const float* x     = (const float*)d_in[0];
    const float* Wgate = (const float*)d_in[1];
    const float* Wg    = (const float*)d_in[2];
    const float* Wu    = (const float*)d_in[3];
    const float* Wd    = (const float*)d_in[4];
    float* out = (float*)d_out;

    uint8_t* wsb = (uint8_t*)d_ws;
    int*   counts  = (int*)wsb;                                    // 32 ints
    int*   tok_ids = (int*)(wsb + 256);                            // E*N ints  (1 MiB)
    float* tok_w   = (float*)(wsb + 256 + (size_t)Ee * Nn * 4);    // E*N float (1 MiB)
    const size_t wbytes = (size_t)Ee * Hh * Ii * sizeof(__bf16);   // 128 MiB each
    __bf16* WgT = (__bf16*)(wsb + (4u << 20));
    __bf16* WuT = (__bf16*)(wsb + (4u << 20) + wbytes);
    __bf16* WdT = (__bf16*)(wsb + (4u << 20) + 2 * wbytes);

    hipMemsetAsync(counts, 0, 256, stream);
    hipMemsetAsync(d_out, 0, (size_t)out_size * sizeof(float), stream);

    // Weight prep: transpose + fp32->bf16 (every launch; deterministic)
    {
        dim3 g1(Ii / 32, Hh / 32, Ee);   // Wg/Wu: [H][I] -> [I][H]
        transpose_cvt_kernel<<<g1, 256, 0, stream>>>(Wg, WgT, Hh, Ii);
        transpose_cvt_kernel<<<g1, 256, 0, stream>>>(Wu, WuT, Hh, Ii);
        dim3 g2(Hh / 32, Ii / 32, Ee);   // Wd: [I][H] -> [H][I]
        transpose_cvt_kernel<<<g2, 256, 0, stream>>>(Wd, WdT, Ii, Hh);
    }

    router_kernel<<<Nn / 8, 256, 0, stream>>>(x, Wgate, counts, tok_ids, tok_w);

    dim3 grid(Nn / 16, Ee);   // x = tile-in-expert (expert-major L2 reuse)
    moe_expert_kernel<<<grid, 256, 0, stream>>>(x, WgT, WuT, WdT,
                                                counts, tok_ids, tok_w, out);
}
